// ATD_CA_57844619542562
// MI455X (gfx1250) — matrix operand order repaired and checked by reading, not compiled, not run on hardware
//
#include <hip/hip_runtime.h>
#include <hip/hip_bf16.h>

#define DIM   192
#define RC    10
#define MTOK  64
#define BATCH 16
#define NTOK  16384
#define LOGM  4.1588830833596715f   // ln(64)

typedef _Float16 v8h  __attribute__((ext_vector_type(8)));
typedef _Float16 v16h __attribute__((ext_vector_type(16)));
typedef float    v8f  __attribute__((ext_vector_type(8)));

static __device__ __forceinline__ v16h cat8(v8h lo, v8h hi) {
  return __builtin_shufflevector(lo, hi, 0,1,2,3,4,5,6,7,8,9,10,11,12,13,14,15);
}
static __device__ __forceinline__ float dot4(float4 a, float4 b) {
  return fmaf(a.x, b.x, fmaf(a.y, b.y, fmaf(a.z, b.z, a.w * b.w)));
}

// ---------------------------------------------------------------------------
// Kernel 1: per batch, compute normalized keys kn (64x10 f32) and
// v = td @ wv^T + b stored TRANSPOSED as f16: vT[d][m]  (d-major, m contiguous)
// ---------------------------------------------------------------------------
__global__ __launch_bounds__(256) void ATD_prep_kernel(
    const float* __restrict__ td, const float* __restrict__ wk_w,
    const float* __restrict__ wk_b, const float* __restrict__ wv_w,
    const float* __restrict__ wv_b, _Float16* __restrict__ vT_ws,
    float* __restrict__ kn_ws)
{
  __shared__ __align__(16) float td_s[MTOK * DIM];
  __shared__ __align__(16) float k_s[MTOK * RC];
  const int tid = threadIdx.x;
  const int b   = blockIdx.x;

  {
    const float4* src = (const float4*)(td + (size_t)b * MTOK * DIM);
    float4* dst = (float4*)td_s;
    for (int i = tid; i < MTOK * DIM / 4; i += 256) dst[i] = src[i];
  }
  __syncthreads();

  // k = td @ wk^T + b
  for (int idx = tid; idx < MTOK * RC; idx += 256) {
    const int m = idx / RC, r = idx - m * RC;
    const float4* w4 = (const float4*)(wk_w + r * DIM);
    const float4* t4 = (const float4*)(td_s + m * DIM);
    float acc = 0.f;
#pragma unroll 8
    for (int i = 0; i < DIM / 4; ++i) acc += dot4(t4[i], w4[i]);
    k_s[idx] = acc + wk_b[r];
  }

  // v[m][d] -> vT[d*64 + m] as f16
  for (int idx = tid; idx < DIM * MTOK; idx += 256) {
    const int d = idx >> 6, m = idx & 63;
    const float4* w4 = (const float4*)(wv_w + d * DIM);
    const float4* t4 = (const float4*)(td_s + m * DIM);
    float acc = 0.f;
#pragma unroll 8
    for (int i = 0; i < DIM / 4; ++i) acc += dot4(t4[i], w4[i]);
    vT_ws[(size_t)b * DIM * MTOK + idx] = (_Float16)(acc + wv_b[d]);
  }
  __syncthreads();

  // normalize keys
  if (tid < MTOK) {
    float ss = 0.f;
#pragma unroll
    for (int r = 0; r < RC; ++r) { float kv = k_s[tid * RC + r]; ss = fmaf(kv, kv, ss); }
    const float inv = 1.0f / fmaxf(sqrtf(ss), 1e-12f);
#pragma unroll
    for (int r = 0; r < RC; ++r)
      kn_ws[(size_t)b * MTOK * RC + tid * RC + r] = k_s[tid * RC + r] * inv;
  }
}

// ---------------------------------------------------------------------------
// Kernel 2: main. 256 threads = 8 waves; each wave owns 16 tokens.
// q proj -> normalize -> logits -> softmax -> attn out + f16 pack -> WMMA attn@v
// ---------------------------------------------------------------------------
struct __align__(16) SMem {
  float    wq[RC * DIM];        // 7680 B
  float    wqb[16];             // 64 B
  float    kn[MTOK * RC];       // 2560 B
  float    sf[MTOK];            // 256 B
  _Float16 vT[DIM * MTOK];      // 24576 B
  _Float16 attn[8 * 16 * MTOK]; // 16384 B
};

__global__ __launch_bounds__(256) void ATD_attn_kernel(
    const float* __restrict__ x, const float* __restrict__ wq_w,
    const float* __restrict__ wq_b, const float* __restrict__ scale,
    const _Float16* __restrict__ vT_ws, const float* __restrict__ kn_ws,
    float* __restrict__ out, float* __restrict__ attn_out)
{
  __shared__ SMem s;
  const int tid     = threadIdx.x;
  const int b       = blockIdx.x >> 7;     // NTOK/128 = 128 tiles per batch
  const int tile    = blockIdx.x & 127;
  const int tokbase = tile * 128;

  // ---- cooperative preload of per-batch constants ----
  for (int i = tid; i < RC * DIM; i += 256) s.wq[i] = wq_w[i];
  if (tid < RC)   s.wqb[tid] = wq_b[tid];
  for (int i = tid; i < MTOK * RC; i += 256) s.kn[i] = kn_ws[b * MTOK * RC + i];
  if (tid < MTOK) s.sf[tid] = 1.0f + fminf(fmaxf(scale[tid], 0.f), 1.f) * LOGM;
  {
    const uint4* src = (const uint4*)(vT_ws + (size_t)b * DIM * MTOK);
    uint4* dst = (uint4*)s.vT;
    for (int i = tid; i < DIM * MTOK * 2 / 16; i += 256) dst[i] = src[i];
  }
  __syncthreads();

  const int wave = tid >> 5;
  const int lane = tid & 31;
  const int t    = lane & 15;   // token within wave / WMMA row or col
  const int hf   = lane >> 4;   // which half-lane group
  const int tok  = tokbase + wave * 16 + t;

  // ---- q projection: lane pair splits DIM in half ----
  float acc[RC];
#pragma unroll
  for (int r = 0; r < RC; ++r) acc[r] = 0.f;
  const float4* x4 = (const float4*)(x + ((size_t)b * NTOK + tok) * DIM + hf * 96);
#pragma unroll 4
  for (int i = 0; i < 24; ++i) {
    const float4 xv = x4[i];
#pragma unroll
    for (int r = 0; r < RC; ++r) {
      const float4 wv = *(const float4*)(&s.wq[r * DIM + hf * 96 + i * 4]);
      acc[r] = fmaf(xv.x, wv.x, fmaf(xv.y, wv.y, fmaf(xv.z, wv.z, fmaf(xv.w, wv.w, acc[r]))));
    }
  }
  float ss = 0.f;
#pragma unroll
  for (int r = 0; r < RC; ++r) {
    acc[r] += __shfl_xor(acc[r], 16, 32);
    acc[r] += s.wqb[r];
    ss = fmaf(acc[r], acc[r], ss);
  }
  const float qinv = 1.0f / fmaxf(sqrtf(ss), 1e-12f);

  // ---- logits: 32 m-values per lane ----
  float p[32];
  const int mbase = hf * 32;
#pragma unroll 4
  for (int j = 0; j < 32; ++j) {
    float d = 0.f;
#pragma unroll
    for (int r = 0; r < RC; ++r) d = fmaf(acc[r], s.kn[(mbase + j) * RC + r], d);
    p[j] = d * qinv * s.sf[mbase + j];
  }

  // ---- softmax over 64 (split across lane pair) ----
  float mx = p[0];
#pragma unroll
  for (int j = 1; j < 32; ++j) mx = fmaxf(mx, p[j]);
  mx = fmaxf(mx, __shfl_xor(mx, 16, 32));
  float sum = 0.f;
#pragma unroll
  for (int j = 0; j < 32; ++j) { p[j] = __expf(p[j] - mx); sum += p[j]; }
  sum += __shfl_xor(sum, 16, 32);
  const float rs = 1.0f / sum;
#pragma unroll
  for (int j = 0; j < 32; ++j) p[j] *= rs;

  // ---- attn output (second region of d_out) ----
  {
    float4* ao = (float4*)(attn_out + ((size_t)b * NTOK + tok) * MTOK + mbase);
#pragma unroll
    for (int g = 0; g < 8; ++g)
      ao[g] = make_float4(p[4 * g], p[4 * g + 1], p[4 * g + 2], p[4 * g + 3]);
  }

  // ---- pack attn to f16 in LDS: attn[wave][token][m] ----
  {
    _Float16* as = s.attn + wave * 1024 + t * 64 + mbase;
#pragma unroll
    for (int g = 0; g < 4; ++g) {
      v8h hv;
#pragma unroll
      for (int e = 0; e < 8; ++e) hv[e] = (_Float16)p[g * 8 + e];
      *(v8h*)(as + g * 8) = hv;
    }
  }
  // cross-lane LDS dependency within the wave: DS ops are in-order per wave;
  // clobber blocks compile-time reordering, wait drains the stores.
  asm volatile("s_wait_dscnt 0" ::: "memory");

  // ---- A tiles (16x32 f16): ISA layout lane<16 K{0..7,16..23}, lane>=16 K{8..15,24..31}
  const _Float16* abase = s.attn + wave * 1024 + t * 64 + hf * 8;
  const v16h a0 = cat8(*(const v8h*)(abase),      *(const v8h*)(abase + 16));
  const v16h a1 = cat8(*(const v8h*)(abase + 32), *(const v8h*)(abase + 48));

  // ---- 12 N-tiles of v: D(16x16) += A(16x32) * B(32x16), K = m ----
  float* orow = out + ((size_t)b * NTOK + tokbase + wave * 16 + hf * 8) * DIM + t;
#pragma unroll 2
  for (int nt = 0; nt < 12; ++nt) {
    // B layout: lane<16 col N=lane holds K{0..7,16..23}; lane>=16 col N=lane-16 holds K{8..15,24..31}
    const _Float16* vb = s.vT + (nt * 16 + t) * 64 + hf * 8;
    const v16h b0 = cat8(*(const v8h*)(vb),      *(const v8h*)(vb + 16));
    const v16h b1 = cat8(*(const v8h*)(vb + 32), *(const v8h*)(vb + 48));
    v8f c = {};
    c = __builtin_amdgcn_wmma_f32_16x16x32_f16(false, a0, false, b0, (short)0, c, false, false);
    c = __builtin_amdgcn_wmma_f32_16x16x32_f16(false, a1, false, b1, (short)0, c, false, false);
    // D layout: lane group hf, VGPR j -> row M = hf*8 + j, col N = t
#pragma unroll
    for (int j = 0; j < 8; ++j)
      orow[(size_t)j * DIM + nt * 16] = c[j];
  }
}

// ---------------------------------------------------------------------------
extern "C" void kernel_launch(void* const* d_in, const int* in_sizes, int n_in,
                              void* d_out, int out_size, void* d_ws, size_t ws_size,
                              hipStream_t stream) {
  const float* x     = (const float*)d_in[0];
  const float* td    = (const float*)d_in[1];
  const float* wq_w  = (const float*)d_in[2];
  const float* wq_b  = (const float*)d_in[3];
  const float* wk_w  = (const float*)d_in[4];
  const float* wk_b  = (const float*)d_in[5];
  const float* wv_w  = (const float*)d_in[6];
  const float* wv_b  = (const float*)d_in[7];
  const float* scale = (const float*)d_in[8];
  (void)in_sizes; (void)n_in; (void)out_size; (void)ws_size;

  float* out      = (float*)d_out;
  float* attn_out = out + (size_t)BATCH * NTOK * DIM;

  _Float16* vT_ws = (_Float16*)d_ws;
  float*    kn_ws = (float*)((char*)d_ws + (size_t)BATCH * DIM * MTOK * sizeof(_Float16));

  ATD_prep_kernel<<<BATCH, 256, 0, stream>>>(td, wk_w, wk_b, wv_w, wv_b, vT_ws, kn_ws);
  ATD_attn_kernel<<<BATCH * (NTOK / 128), 256, 0, stream>>>(
      x, wq_w, wq_b, scale, vT_ws, kn_ws, out, attn_out);
}
